// MultiHeadAttention_47863115546801
// MI455X (gfx1250) — matrix operand order repaired and checked by reading, not compiled, not run on hardware
//
#include <hip/hip_runtime.h>
#include <hip/hip_bf16.h>
#include <math.h>
#include <stdint.h>

#define D_MODEL 1024
#define NHEAD   16
#define D_HEAD  64
#define BATCH   2
#define SEQ     2048
#define MTOT    (BATCH*SEQ)   /* 4096 rows total */

typedef __bf16 bf16;
typedef __attribute__((ext_vector_type(16))) __bf16 bf16x16;
typedef __attribute__((ext_vector_type(8)))  __bf16 bf16x8;
typedef __attribute__((ext_vector_type(4)))  __bf16 bf16x4;
typedef __attribute__((ext_vector_type(8)))  float  f32x8;
typedef __attribute__((ext_vector_type(4)))  unsigned int u32x4;
typedef __attribute__((ext_vector_type(8)))  int i32x8;
typedef __attribute__((ext_vector_type(4)))  int i32x4;

#if __has_builtin(__builtin_amdgcn_tensor_load_to_lds) && \
    __has_builtin(__builtin_amdgcn_s_wait_tensorcnt)
#define HAVE_TDM 1
#endif

// ---------------------------------------------------------------------------
// Tensor Data Mover: 2D bf16 tile (tile1 rows x tile0 elems) from a strided
// global tensor into LDS (rows packed contiguously). D# layout per CDNA5 ISA
// ch.8: group0 = {count/type, lds_addr, global_addr}, group1 = {data_size,
// tensor dims, tile dims, dim0 stride}. Issued by one wave; EXEC is ignored.
// ---------------------------------------------------------------------------
#ifdef HAVE_TDM
__device__ __forceinline__ void tdm_load_2d_bf16(const bf16* gptr, uint32_t lds_byte,
                                                 uint32_t td0, uint32_t td1,
                                                 uint64_t stride0_elems,
                                                 uint32_t tile0, uint32_t tile1) {
  uint64_t ga = (uint64_t)(uintptr_t)gptr;
  u32x4 g0;
  g0[0] = 1u;                                    // count=1, user descriptor
  g0[1] = lds_byte;                              // D#.lds_addr
  g0[2] = (uint32_t)ga;                          // global_addr[31:0]
  g0[3] = ((uint32_t)(ga >> 32) & 0x01FFFFFFu)   // global_addr[56:32]
          | 0x80000000u;                         // type=2 ("image")
  i32x8 g1;
  g1[0] = 0x00010000;                            // data_size=1 (2 bytes)
  g1[1] = (int)((td0 & 0xFFFFu) << 16);          // tensor_dim0[15:0]
  g1[2] = (int)((td0 >> 16) | ((td1 & 0xFFFFu) << 16));   // td0 hi | td1 lo
  g1[3] = (int)((td1 >> 16) | (tile0 << 16));    // td1 hi | tile_dim0
  g1[4] = (int)(tile1 & 0xFFFFu);                // tile_dim1 (tile_dim2=0)
  g1[5] = (int)(uint32_t)stride0_elems;          // tensor_dim0_stride[31:0]
  g1[6] = (int)((uint32_t)(stride0_elems >> 32) & 0xFFFFu);
  g1[7] = 0;
  i32x4 z4 = {};
#if defined(__clang_major__) && (__clang_major__ >= 23)
  i32x8 z8 = {};
  __builtin_amdgcn_tensor_load_to_lds(g0, g1, z4, z4, z8, 0);
#else
  __builtin_amdgcn_tensor_load_to_lds(g0, g1, z4, z4, 0);
#endif
}
#endif

// ---------------------------------------------------------------------------
// WMMA fragment loaders (CDNA5 V_WMMA_F32_16X16X32_BF16, wave32).
// ---------------------------------------------------------------------------
__device__ __forceinline__ bf16x16 frag_ldA(const bf16* p) {
  bf16x8 lo = *(const bf16x8*)(p);
  bf16x8 hi = *(const bf16x8*)(p + 16);
  return __builtin_shufflevector(lo, hi, 0,1,2,3,4,5,6,7,8,9,10,11,12,13,14,15);
}
__device__ __forceinline__ bf16x16 frag_ldB(const bf16* p) {
  bf16x8 lo = *(const bf16x8*)(p);
  bf16x8 hi = *(const bf16x8*)(p + 16);
  return __builtin_shufflevector(lo, hi, 0,1,2,3,4,5,6,7,8,9,10,11,12,13,14,15);
}
__device__ __forceinline__ f32x8 wmma_bf16(bf16x16 a, bf16x16 b, f32x8 c) {
  return __builtin_amdgcn_wmma_f32_16x16x32_bf16(false, a, false, b,
                                                 (short)0, c, false, false);
}

// ---------------------------------------------------------------------------
// fp32 -> bf16 conversion (grid-stride, 4-wide)
// ---------------------------------------------------------------------------
__global__ __launch_bounds__(256) void cvt_f32_bf16(const float* __restrict__ s,
                                                    bf16* __restrict__ d, int n) {
  int i = (blockIdx.x * blockDim.x + threadIdx.x) * 4;
  int stride = gridDim.x * blockDim.x * 4;
  for (; i < n; i += stride) {
    float4 f = *(const float4*)(s + i);
    bf16x4 o = { (bf16)f.x, (bf16)f.y, (bf16)f.z, (bf16)f.w };
    *(bf16x4*)(d + i) = o;
  }
}

// ---------------------------------------------------------------------------
// GEMM: C[M,N] = A[M,K] * W[N,K]^T   (A,W bf16; M=4096,N=K=1024)
// Block 256 thr = 8 waves; tile 128(M) x 64(N); wave w -> 16 rows, 64 cols.
// Tiles staged into LDS by the TDM when available, else manual copy.
// ---------------------------------------------------------------------------
template<int OUTMODE>   // 0: bf16 row-major out, 1: f32 row-major out
__global__ __launch_bounds__(256) void gemm_wmma(const bf16* __restrict__ A,
                                                 const bf16* __restrict__ W,
                                                 void* __restrict__ out) {
  __shared__ __align__(64) bf16 As[128 * 32];
  __shared__ __align__(64) bf16 Ws[64 * 32];
  const int tid  = threadIdx.x;
  const int lane = tid & 31;
  const int wv   = tid >> 5;       // 0..7
  const int hi   = lane >> 4;      // half-wave
  const int ln   = lane & 15;
  const int m0 = blockIdx.y * 128;
  const int n0 = blockIdx.x * 64;

  f32x8 acc[4] = {};

  for (int kt = 0; kt < D_MODEL / 32; ++kt) {
    __syncthreads();
#ifdef HAVE_TDM
    if (wv == 0) {
      tdm_load_2d_bf16(&A[(size_t)m0 * D_MODEL + kt * 32],
                       (uint32_t)(uintptr_t)&As[0], 32, 128, D_MODEL, 32, 128);
      tdm_load_2d_bf16(&W[(size_t)n0 * D_MODEL + kt * 32],
                       (uint32_t)(uintptr_t)&Ws[0], 32, 64, D_MODEL, 32, 64);
      __builtin_amdgcn_s_wait_tensorcnt(0);
    }
#else
    #pragma unroll
    for (int t = 0; t < 2; ++t) {
      int c = tid + t * 256;
      int r = c >> 2, off = (c & 3) * 8;
      *(uint4*)(&As[r * 32 + off]) =
          *(const uint4*)(&A[(size_t)(m0 + r) * D_MODEL + kt * 32 + off]);
    }
    {
      int r = tid >> 2, off = (tid & 3) * 8;
      *(uint4*)(&Ws[r * 32 + off]) =
          *(const uint4*)(&W[(size_t)(n0 + r) * D_MODEL + kt * 32 + off]);
    }
#endif
    __syncthreads();

    bf16x16 af = frag_ldA(&As[(wv * 16 + ln) * 32 + hi * 8]);
    #pragma unroll
    for (int a = 0; a < 4; ++a) {
      bf16x16 bfr = frag_ldB(&Ws[(a * 16 + ln) * 32 + hi * 8]);
      acc[a] = wmma_bf16(af, bfr, acc[a]);
    }
  }

  // C/D layout: lane L -> col n=L&15, rows m = 8*(L>>4)+r for VGPR r
  #pragma unroll
  for (int a = 0; a < 4; ++a) {
    int n = n0 + a * 16 + ln;
    #pragma unroll
    for (int r = 0; r < 8; ++r) {
      int m = m0 + wv * 16 + hi * 8 + r;
      if (OUTMODE == 0)
        ((bf16*)out)[(size_t)m * D_MODEL + n] = (bf16)acc[a][r];
      else
        ((float*)out)[(size_t)m * D_MODEL + n] = acc[a][r];
    }
  }
}

// ---------------------------------------------------------------------------
// Flash attention over one (b,h). K tile staged by TDM; V transposed into
// LDS manually (TDM cannot transpose); P round-trips through LDS to convert
// accumulator layout into A-fragment layout.
// ---------------------------------------------------------------------------
__global__ __launch_bounds__(128) void flash_attn(const bf16* __restrict__ Qh,
                                                  const bf16* __restrict__ Kh,
                                                  const bf16* __restrict__ Vh,
                                                  const int*  __restrict__ mask,
                                                  bf16* __restrict__ AO) {
  __shared__ __align__(64) bf16 Ksm[64 * 64];
  __shared__ __align__(64) bf16 VT [64 * 64];
  __shared__ __align__(64) bf16 Psm[4 * 16 * 64];

  const int tid = threadIdx.x, lane = tid & 31, wv = tid >> 5;
  const int hi = lane >> 4, ln = lane & 15;
  const int bh = blockIdx.x;               // 0..B*H-1
  const int b  = bh / NHEAD;
  const int q0 = blockIdx.y * 64 + wv * 16;

  const bf16* Qb = Qh + (size_t)bh * SEQ * D_HEAD;
  const bf16* Kb = Kh + (size_t)bh * SEQ * D_HEAD;
  const bf16* Vb = Vh + (size_t)bh * SEQ * D_HEAD;

  // Q fragments (16 rows x 64 d) loaded once, straight from global
  bf16x16 qf[2];
  #pragma unroll
  for (int ks = 0; ks < 2; ++ks)
    qf[ks] = frag_ldA(Qb + (size_t)(q0 + ln) * D_HEAD + ks * 32 + hi * 8);

  float mrow[8], lrow[8];
  f32x8 oacc[4] = {};
  #pragma unroll
  for (int r = 0; r < 8; ++r) { mrow[r] = -3.0e38f; lrow[r] = 0.f; }

  for (int kt = 0; kt < SEQ / 64; ++kt) {
    __syncthreads();
#ifdef HAVE_TDM
    if (wv == 0)
      tdm_load_2d_bf16(Kb + (size_t)kt * 64 * D_HEAD,
                       (uint32_t)(uintptr_t)&Ksm[0], 64, 64, D_HEAD, 64, 64);
#else
    #pragma unroll
    for (int t = 0; t < 4; ++t) {
      int c = tid + t * 128;
      int r = c >> 3, off = (c & 7) * 8;
      *(uint4*)(&Ksm[r * 64 + off]) =
          *(const uint4*)(&Kb[(size_t)(kt * 64 + r) * D_HEAD + off]);
    }
#endif
    // stage V transposed: VT[d][key]
    #pragma unroll
    for (int t = 0; t < 4; ++t) {
      int c = tid + t * 128;
      int kk = c >> 3, d0 = (c & 7) * 8;
      bf16x8 vv = *(const bf16x8*)(&Vb[(size_t)(kt * 64 + kk) * D_HEAD + d0]);
      #pragma unroll
      for (int j = 0; j < 8; ++j) VT[(d0 + j) * 64 + kk] = vv[j];
    }
#ifdef HAVE_TDM
    if (wv == 0) __builtin_amdgcn_s_wait_tensorcnt(0);
#endif
    __syncthreads();

    // S = Q K^T : 4 accs cover 64 keys, K-dim 64 = 2 wmma steps
    f32x8 sc[4] = {};
    #pragma unroll
    for (int a = 0; a < 4; ++a) {
      #pragma unroll
      for (int ks = 0; ks < 2; ++ks) {
        bf16x16 bfr = frag_ldB(&Ksm[(a * 16 + ln) * 64 + ks * 32 + hi * 8]);
        sc[a] = wmma_bf16(qf[ks], bfr, sc[a]);
      }
    }

    // scale + boolean mask (mask true -> -inf)
    #pragma unroll
    for (int a = 0; a < 4; ++a) {
      int kg = kt * 64 + a * 16 + ln;
      const int* mp = mask + (size_t)b * SEQ * SEQ + kg;
      #pragma unroll
      for (int r = 0; r < 8; ++r) {
        int qg = q0 + hi * 8 + r;
        float s = sc[a][r] * 0.125f;       // 1/sqrt(64)
        if (mp[(size_t)qg * SEQ]) s = -3.0e38f;
        sc[a][r] = s;
      }
    }

    // online softmax: per-lane stats for its 8 rows, reduced across the 16
    // lanes of each half-wave with shfl_xor (wave32)
    #pragma unroll
    for (int r = 0; r < 8; ++r) {
      float cm = fmaxf(fmaxf(sc[0][r], sc[1][r]), fmaxf(sc[2][r], sc[3][r]));
      #pragma unroll
      for (int off = 1; off < 16; off <<= 1)
        cm = fmaxf(cm, __shfl_xor(cm, off, 32));
      float mnew  = fmaxf(mrow[r], cm);
      float alpha = __expf(mrow[r] - mnew);
      float ps = 0.f;
      #pragma unroll
      for (int a = 0; a < 4; ++a) {
        float p = __expf(sc[a][r] - mnew);
        sc[a][r] = p; ps += p;
      }
      #pragma unroll
      for (int off = 1; off < 16; off <<= 1)
        ps += __shfl_xor(ps, off, 32);
      lrow[r] = lrow[r] * alpha + ps;
      mrow[r] = mnew;
      #pragma unroll
      for (int a = 0; a < 4; ++a) oacc[a][r] *= alpha;
    }

    // P tile (C-layout) -> LDS -> A-fragment layout
    bf16* Pw = &Psm[wv * 16 * 64];
    #pragma unroll
    for (int a = 0; a < 4; ++a)
      #pragma unroll
      for (int r = 0; r < 8; ++r)
        Pw[(hi * 8 + r) * 64 + a * 16 + ln] = (bf16)sc[a][r];
    __syncthreads();

    // O += P V : A = P (16x64), B = V via VT rows (contiguous B-fragments)
    bf16x16 pf[2];
    #pragma unroll
    for (int ks = 0; ks < 2; ++ks)
      pf[ks] = frag_ldA(&Pw[ln * 64 + ks * 32 + hi * 8]);
    #pragma unroll
    for (int a = 0; a < 4; ++a) {
      #pragma unroll
      for (int ks = 0; ks < 2; ++ks) {
        bf16x16 vf = frag_ldB(&VT[(a * 16 + ln) * 64 + ks * 32 + hi * 8]);
        oacc[a] = wmma_bf16(pf[ks], vf, oacc[a]);
      }
    }
  }

  // normalize and store (row-major (B*H, SEQ, 64) == reference reshape)
  bf16* AOb = AO + (size_t)bh * SEQ * D_HEAD;
  #pragma unroll
  for (int r = 0; r < 8; ++r) {
    float inv = lrow[r] > 0.f ? 1.f / lrow[r] : 0.f;
    int qg = q0 + hi * 8 + r;
    #pragma unroll
    for (int a = 0; a < 4; ++a)
      AOb[(size_t)qg * D_HEAD + a * 16 + ln] = (bf16)(oacc[a][r] * inv);
  }
}

// ---------------------------------------------------------------------------
extern "C" void kernel_launch(void* const* d_in, const int* in_sizes, int n_in,
                              void* d_out, int out_size, void* d_ws, size_t ws_size,
                              hipStream_t stream) {
  (void)in_sizes; (void)n_in; (void)out_size; (void)ws_size;
  const float* q    = (const float*)d_in[0];
  const float* k    = (const float*)d_in[1];
  const float* v    = (const float*)d_in[2];
  const int*   mask = (const int*)  d_in[3];
  const float* wq   = (const float*)d_in[4];
  const float* wk   = (const float*)d_in[5];
  const float* wv   = (const float*)d_in[6];
  const float* wo   = (const float*)d_in[7];

  const size_t WSZ = (size_t)D_MODEL * D_MODEL;  // 1M elems
  const size_t ASZ = (size_t)MTOT * D_MODEL;     // 4M elems
  bf16* wq_b = (bf16*)d_ws;
  bf16* wk_b = wq_b + WSZ;
  bf16* wv_b = wk_b + WSZ;
  bf16* wo_b = wv_b + WSZ;
  bf16* q_b  = wo_b + WSZ;
  bf16* k_b  = q_b + ASZ;
  bf16* v_b  = k_b + ASZ;
  bf16* qh   = v_b + ASZ;
  bf16* kh   = qh + ASZ;
  bf16* vh   = kh + ASZ;
  bf16* ao   = vh + ASZ;   // total ws use = 64 MB

  cvt_f32_bf16<<<1024, 256, 0, stream>>>(wq, wq_b, (int)WSZ);
  cvt_f32_bf16<<<1024, 256, 0, stream>>>(wk, wk_b, (int)WSZ);
  cvt_f32_bf16<<<1024, 256, 0, stream>>>(wv, wv_b, (int)WSZ);
  cvt_f32_bf16<<<1024, 256, 0, stream>>>(wo, wo_b, (int)WSZ);
  cvt_f32_bf16<<<2048, 256, 0, stream>>>(q, q_b, (int)ASZ);
  cvt_f32_bf16<<<2048, 256, 0, stream>>>(k, k_b, (int)ASZ);
  cvt_f32_bf16<<<2048, 256, 0, stream>>>(v, v_b, (int)ASZ);

  dim3 ggrid(D_MODEL / 64, MTOT / 128);
  gemm_wmma<0><<<ggrid, 256, 0, stream>>>(q_b, wq_b, qh);
  gemm_wmma<0><<<ggrid, 256, 0, stream>>>(k_b, wk_b, kh);
  gemm_wmma<0><<<ggrid, 256, 0, stream>>>(v_b, wv_b, vh);

  flash_attn<<<dim3(BATCH * NHEAD, SEQ / 64), 128, 0, stream>>>(qh, kh, vh, mask, ao);

  gemm_wmma<1><<<ggrid, 256, 0, stream>>>(ao, wo_b, d_out);
}
